// TwoLayerResidualGAT_429496730274
// MI455X (gfx1250) — matrix operand order repaired and checked by reading, not compiled, not run on hardware
//
#include <hip/hip_runtime.h>
#include <hip/hip_bf16.h>
#include <math.h>

// ---------------------------------------------------------------------------
// Two-layer residual GAT for MI455X (gfx1250, wave32, WMMA).
// GEMMs: bf16 hi/lo split (3x v_wmma_f32_16x16x32_bf16 per K-step) ~ fp32 acc,
//        with the shared B tile staged to LDS via async DMA
//        (global_load_async_to_lds_b128, double-buffered, ASYNCcnt-tracked).
// Edge phase: atomic segment-max/sum softmax + per-wave gather/atomic scatter.
// ---------------------------------------------------------------------------

typedef __attribute__((ext_vector_type(16))) __bf16 v16bf;
typedef __attribute__((ext_vector_type(8)))  float  v8f;

static __device__ __forceinline__ void edge_ep(const int* __restrict__ ei, int E,
                                               int e, int& s, int& d) {
  if (e < E) { s = ei[e]; d = ei[E + e]; }
  else       { s = e - E; d = e - E; }          // self-loops appended
}

// float atomic max via signed/unsigned int monotonicity trick
static __device__ __forceinline__ void atomic_max_f32(float* addr, float v) {
  if (v >= 0.f) atomicMax((int*)addr,          __float_as_int(v));
  else          atomicMin((unsigned int*)addr, __float_as_uint(v));
}

// async copy: global -> LDS, 16 bytes, tracked by ASYNCcnt (ISA 08 §4)
static __device__ __forceinline__ void async_b128(unsigned lds_off,
                                                  const void* gptr) {
  asm volatile("global_load_async_to_lds_b128 %0, %1, off"
               :: "v"(lds_off), "v"((unsigned long long)(uintptr_t)gptr)
               : "memory");
}

// ---------------------------------------------------------------------------
__global__ void fill_f32(float* __restrict__ p, long n, float v) {
  long i = (long)blockIdx.x * blockDim.x + threadIdx.x;
  if (i < n) p[i] = v;
}

// Transpose W[K][N] -> Wt[N][K], split fp32 into bf16 hi + lo residual.
__global__ void prep_w(const float* __restrict__ W, __bf16* __restrict__ Th,
                       __bf16* __restrict__ Tl, int K, int Ncols) {
  long i = (long)blockIdx.x * blockDim.x + threadIdx.x;
  if (i >= (long)K * Ncols) return;
  int k = (int)(i / Ncols), n = (int)(i % Ncols);
  float w = W[i];
  __bf16 h = (__bf16)w;
  Th[(long)n * K + k] = h;
  Tl[(long)n * K + k] = (__bf16)(w - (float)h);
}

// ---------------------------------------------------------------------------
// GEMM: C[M][Ncols] = A[M][K] * Bt^T  (Bt is [Ncols][K] bf16 hi/lo).
// Block: 256 thr = 8 waves; wave w -> 16 rows x 64 cols (4 N-subtiles).
// B tile (64 cols x 32 K, hi+lo = 8KB) is shared by all 8 waves: staged to LDS
// with async DMA, double buffered; DMA of step s+1 overlaps WMMAs of step s.
// mode 0: C = AB ; mode 1: C = elu(AB + R + b0 + b1) ; mode 2: C = AB + R + b0 + b1
// Requires: K % 32 == 0, Ncols % 64 == 0.
__global__ __launch_bounds__(256) void gemm_bf16x3(
    const float* __restrict__ A, const __bf16* __restrict__ Bth,
    const __bf16* __restrict__ Btl, float* __restrict__ C,
    const float* __restrict__ R, const float* __restrict__ bias0,
    const float* __restrict__ bias1, int M, int K, int Ncols, int mode)
{
  __shared__ __align__(16) __bf16 sBh[2][64 * 32];   // [buf][col*32 + k]
  __shared__ __align__(16) __bf16 sBl[2][64 * 32];

  const int tid   = threadIdx.x;
  const int lane  = tid & 31;
  const int wave  = tid >> 5;
  const int tileM = blockIdx.y * 8 + wave;
  const bool valid = (tileM * 16 < M);         // wave-uniform
  const int nbase  = blockIdx.x * 64;
  const int half   = lane >> 4;
  const int l16    = lane & 15;
  int mrow = tileM * 16 + l16;
  if (mrow >= M) mrow = M - 1;                 // clamp: tail waves compute garbage,
                                               // keep EXEC full for WMMA, skip store
  const int kbaseA = half * 16;                // A: contiguous 16 K values per half
  const int kbaseB = half * 16;                // B: contiguous 16 K values per half

  // staging assignment: thread t loads 8 bf16 (16B) of hi and of lo
  const int sc = tid >> 2;                     // column 0..63
  const int sq = tid & 3;                      // K-quarter 0..3
  const size_t gb = (size_t)(nbase + sc) * K + sq * 8;
  const unsigned ldsoff = (unsigned)(sc * 32 + sq * 8);

  const int S = K / 32;

  v8f acc[4];
  #pragma unroll
  for (int s = 0; s < 4; ++s) acc[s] = (v8f){0.f,0.f,0.f,0.f,0.f,0.f,0.f,0.f};

  const float* Arow = A + (size_t)mrow * K;

  // prologue: DMA B tile for step 0
  {
    unsigned dh = (unsigned)(uintptr_t)&sBh[0][0] + ldsoff * 2u;
    unsigned dl = (unsigned)(uintptr_t)&sBl[0][0] + ldsoff * 2u;
    async_b128(dh, Bth + gb);
    async_b128(dl, Btl + gb);
  }

  for (int s = 0; s < S; ++s) {
    const int k0 = s * 32;

    // ---- A fragment: fp32 -> bf16 hi/lo split (private, no sharing) ----
    const float4 a0 = *(const float4*)(Arow + k0 + kbaseA);
    const float4 a1 = *(const float4*)(Arow + k0 + kbaseA + 4);
    const float4 a2 = *(const float4*)(Arow + k0 + kbaseA + 8);
    const float4 a3 = *(const float4*)(Arow + k0 + kbaseA + 12);
    float af[16] = { a0.x,a0.y,a0.z,a0.w, a1.x,a1.y,a1.z,a1.w,
                     a2.x,a2.y,a2.z,a2.w, a3.x,a3.y,a3.z,a3.w };
    v16bf ahi, alo;
    #pragma unroll
    for (int e = 0; e < 16; ++e) {
      __bf16 h = (__bf16)af[e];
      ahi[e] = h;
      alo[e] = (__bf16)(af[e] - (float)h);
    }
    if (k0 + 32 < K) __builtin_prefetch(Arow + k0 + 32 + kbaseA, 0, 3);

    // ---- kick DMA for next step into the other buffer, retire current ----
    if (s + 1 < S) {
      const size_t gnext = gb + (size_t)(s + 1) * 32;
      unsigned dh = (unsigned)(uintptr_t)&sBh[(s + 1) & 1][0] + ldsoff * 2u;
      unsigned dl = (unsigned)(uintptr_t)&sBl[(s + 1) & 1][0] + ldsoff * 2u;
      async_b128(dh, Bth + gnext);
      async_b128(dl, Btl + gnext);
      asm volatile("s_wait_asynccnt 0x2" ::: "memory");  // current buf landed
    } else {
      asm volatile("s_wait_asynccnt 0x0" ::: "memory");
    }
    __syncthreads();                            // publish LDS across all waves

    const __bf16* bh = &sBh[s & 1][0];
    const __bf16* bl = &sBl[s & 1][0];
    #pragma unroll
    for (int sub = 0; sub < 4; ++sub) {
      const int col = sub * 16 + l16;
      v16bf bhi = *(const v16bf*)(bh + col * 32 + kbaseB);  // 2x ds_load_b128
      v16bf blo = *(const v16bf*)(bl + col * 32 + kbaseB);
      acc[sub] = __builtin_amdgcn_wmma_f32_16x16x32_bf16(false, ahi, false, bhi,
                                                         (short)0, acc[sub], false, false);
      acc[sub] = __builtin_amdgcn_wmma_f32_16x16x32_bf16(false, ahi, false, blo,
                                                         (short)0, acc[sub], false, false);
      acc[sub] = __builtin_amdgcn_wmma_f32_16x16x32_bf16(false, alo, false, bhi,
                                                         (short)0, acc[sub], false, false);
    }
    __syncthreads();                            // all readers done before buf reuse
  }

  if (!valid) return;                           // wave-uniform

  // ---- epilogue (C layout: vreg r -> row = 8*half + r, col = lane&15) ----
  #pragma unroll
  for (int sub = 0; sub < 4; ++sub) {
    const int ncol = nbase + sub * 16 + l16;
    float badd = 0.f;
    if (mode != 0) badd = bias0[ncol] + bias1[ncol];
    #pragma unroll
    for (int r = 0; r < 8; ++r) {
      const int    row = tileM * 16 + half * 8 + r;
      const size_t idx = (size_t)row * Ncols + ncol;
      float v = acc[sub][r];
      if (mode != 0) {
        v += R[idx] + badd;
        if (mode == 1) v = (v > 0.f) ? v : expm1f(v);   // ELU
      }
      C[idx] = v;
    }
  }
}

// ---------------------------------------------------------------------------
// Per-(node, head) attention scores: s = <h[n,h,:], att[h,:]>  (one wave each)
__global__ __launch_bounds__(256) void node_scores(
    const float* __restrict__ Ht, const float* __restrict__ a_src,
    const float* __restrict__ a_dst, float* __restrict__ s_src,
    float* __restrict__ s_dst, int Nn, int H, int D)
{
  const long tid  = (long)blockIdx.x * blockDim.x + threadIdx.x;
  const long wid  = tid >> 5;
  const int  lane = threadIdx.x & 31;
  if (wid >= (long)Nn * H) return;
  const int n = (int)(wid / H), h = (int)(wid % H);
  const float* hp = Ht + ((size_t)n * H + h) * D;
  const float* as = a_src + (size_t)h * D;
  const float* ad = a_dst + (size_t)h * D;
  float ss = 0.f, sd = 0.f;
  for (int d = lane; d < D; d += 32) {
    float hv = hp[d];
    ss += hv * as[d];
    sd += hv * ad[d];
  }
  #pragma unroll
  for (int off = 16; off; off >>= 1) {
    ss += __shfl_xor(ss, off, 32);
    sd += __shfl_xor(sd, off, 32);
  }
  if (lane == 0) { s_src[wid] = ss; s_dst[wid] = sd; }
}

// edge pass 1: e = leaky_relu(s_src[src]+s_dst[dst]); segment-max into mmax
__global__ void edge_pass1(const int* __restrict__ ei, int E, int Et, int H,
                           const float* __restrict__ ssrc,
                           const float* __restrict__ sdst,
                           float* __restrict__ stage, float* __restrict__ mmax)
{
  long t = (long)blockIdx.x * blockDim.x + threadIdx.x;
  if (t >= (long)Et * H) return;
  int e = (int)(t / H), h = (int)(t % H);
  int s, d; edge_ep(ei, E, e, s, d);
  float v = ssrc[(size_t)s * H + h] + sdst[(size_t)d * H + h];
  v = (v >= 0.f) ? v : 0.2f * v;
  stage[t] = v;
  atomic_max_f32(&mmax[(size_t)d * H + h], v);
}

// edge pass 2: p = exp(e - m[dst]); segment-sum into z
__global__ void edge_pass2(const int* __restrict__ ei, int E, int Et, int H,
                           float* __restrict__ stage,
                           const float* __restrict__ mmax, float* __restrict__ z)
{
  long t = (long)blockIdx.x * blockDim.x + threadIdx.x;
  if (t >= (long)Et * H) return;
  int e = (int)(t / H), h = (int)(t % H);
  int s, d; edge_ep(ei, E, e, s, d);
  float p = __expf(stage[t] - mmax[(size_t)d * H + h]);
  stage[t] = p;
  atomicAdd(&z[(size_t)d * H + h], p);
}

// messages: alpha = p/(z+eps); acc[dst,h,:] += alpha * h[src,h,:]  (wave per edge-head)
__global__ __launch_bounds__(256) void scatter_msgs(
    const int* __restrict__ ei, int E, int Et, int H, int D,
    const float* __restrict__ Ht, const float* __restrict__ stage,
    const float* __restrict__ z, float* __restrict__ alpha_out,
    float* __restrict__ acc)
{
  const long tid  = (long)blockIdx.x * blockDim.x + threadIdx.x;
  const long wid  = tid >> 5;
  const int  lane = threadIdx.x & 31;
  if (wid >= (long)Et * H) return;
  int e = (int)(wid / H), h = (int)(wid % H);
  int s, d; edge_ep(ei, E, e, s, d);
  const float a = stage[wid] / (z[(size_t)d * H + h] + 1e-16f);
  if (lane == 0) alpha_out[wid] = a;
  const float* hp = Ht + ((size_t)s * H + h) * D;
  float*       op = acc + ((size_t)d * H + h) * D;
  for (int dd = lane; dd < D; dd += 32)
    atomicAdd(op + dd, hp[dd] * a);              // L2-resident f32 atomics
}

// ---------------------------------------------------------------------------
static inline unsigned cdiv(long a, long b) { return (unsigned)((a + b - 1) / b); }

extern "C" void kernel_launch(void* const* d_in, const int* in_sizes, int n_in,
                              void* d_out, int out_size, void* d_ws, size_t ws_size,
                              hipStream_t stream) {
  const float* x        = (const float*)d_in[0];
  const int*   ei       = (const int*)  d_in[1];
  const float* W1       = (const float*)d_in[2];
  const float* att_src1 = (const float*)d_in[3];
  const float* att_dst1 = (const float*)d_in[4];
  const float* b1       = (const float*)d_in[5];
  const float* lin1_W   = (const float*)d_in[6];
  const float* lin1_b   = (const float*)d_in[7];
  const float* W2       = (const float*)d_in[8];
  const float* att_src2 = (const float*)d_in[9];
  const float* att_dst2 = (const float*)d_in[10];
  const float* b2       = (const float*)d_in[11];
  const float* lin2_W   = (const float*)d_in[12];
  const float* lin2_b   = (const float*)d_in[13];

  const int F = 128, H1n = 4, D1n = 256, D2n = 256;
  const int N   = in_sizes[0] / F;       // 10000
  const int E   = in_sizes[1] / 2;       // 160000
  const int Et  = E + N;                 // + self-loops
  const int HD1 = H1n * D1n;             // 1024

  // ---- workspace bump allocator (256B aligned) ----
  char* wp = (char*)d_ws;
  auto alloc = [&](size_t bytes) -> char* {
    char* p = wp; wp += (bytes + 255) & ~(size_t)255; return p;
  };
  float*  h_t1   = (float*)alloc((size_t)N * HD1 * 4);   // x@W1, later reused as h1
  float*  acc1   = (float*)alloc((size_t)N * HD1 * 4);   // layer-1 scatter accum
  float*  h2t    = (float*)alloc((size_t)N * D2n * 4);   // h1@W2
  float*  acc2   = (float*)alloc((size_t)N * D2n * 4);   // layer-2 scatter accum
  float*  ssrc1  = (float*)alloc((size_t)N * H1n * 4);
  float*  sdst1  = (float*)alloc((size_t)N * H1n * 4);
  float*  m1     = (float*)alloc((size_t)N * H1n * 4);
  float*  z1     = (float*)alloc((size_t)N * H1n * 4);
  float*  ssrc2  = (float*)alloc((size_t)N * 4);
  float*  sdst2  = (float*)alloc((size_t)N * 4);
  float*  m2     = (float*)alloc((size_t)N * 4);
  float*  z2     = (float*)alloc((size_t)N * 4);
  float*  stage1 = (float*)alloc((size_t)Et * H1n * 4);
  float*  stage2 = (float*)alloc((size_t)Et * 4);
  float*  alpha2 = (float*)alloc((size_t)Et * 4);        // unused output, kept in ws
  __bf16* w1h = (__bf16*)alloc((size_t)F * HD1 * 2);
  __bf16* w1l = (__bf16*)alloc((size_t)F * HD1 * 2);
  __bf16* l1h = (__bf16*)alloc((size_t)F * HD1 * 2);
  __bf16* l1l = (__bf16*)alloc((size_t)F * HD1 * 2);
  __bf16* w2h = (__bf16*)alloc((size_t)HD1 * D2n * 2);
  __bf16* w2l = (__bf16*)alloc((size_t)HD1 * D2n * 2);
  __bf16* l2h = (__bf16*)alloc((size_t)HD1 * D2n * 2);
  __bf16* l2l = (__bf16*)alloc((size_t)HD1 * D2n * 2);

  float* out    = (float*)d_out;                         // [N, 256]
  float* alpha1 = out + (size_t)N * D2n;                 // [Et, H1]

  const dim3 B(256);

  // ---- weight prep (transpose + bf16 hi/lo split) ----
  prep_w<<<cdiv((long)F * HD1, 256),   B, 0, stream>>>(W1,     w1h, w1l, F,   HD1);
  prep_w<<<cdiv((long)F * HD1, 256),   B, 0, stream>>>(lin1_W, l1h, l1l, F,   HD1);
  prep_w<<<cdiv((long)HD1 * D2n, 256), B, 0, stream>>>(W2,     w2h, w2l, HD1, D2n);
  prep_w<<<cdiv((long)HD1 * D2n, 256), B, 0, stream>>>(lin2_W, l2h, l2l, HD1, D2n);

  // ---- init accumulators / softmax state ----
  fill_f32<<<cdiv((long)N * HD1, 256), B, 0, stream>>>(acc1, (long)N * HD1, 0.f);
  fill_f32<<<cdiv((long)N * H1n, 256), B, 0, stream>>>(m1,   (long)N * H1n, -INFINITY);
  fill_f32<<<cdiv((long)N * H1n, 256), B, 0, stream>>>(z1,   (long)N * H1n, 0.f);
  fill_f32<<<cdiv((long)N * D2n, 256), B, 0, stream>>>(acc2, (long)N * D2n, 0.f);
  fill_f32<<<cdiv((long)N, 256),       B, 0, stream>>>(m2,   (long)N,       -INFINITY);
  fill_f32<<<cdiv((long)N, 256),       B, 0, stream>>>(z2,   (long)N,       0.f);

  const dim3 gemmGridW(HD1 / 64, cdiv(N / 16, 8));   // N-cols 1024
  const dim3 gemmGridN(D2n / 64, cdiv(N / 16, 8));   // N-cols 256

  // ================= layer 1 =================
  gemm_bf16x3<<<gemmGridW, B, 0, stream>>>(x, w1h, w1l, h_t1,
                                           nullptr, nullptr, nullptr, N, F, HD1, 0);
  node_scores<<<cdiv((long)N * H1n * 32, 256), B, 0, stream>>>(
      h_t1, att_src1, att_dst1, ssrc1, sdst1, N, H1n, D1n);
  edge_pass1<<<cdiv((long)Et * H1n, 256), B, 0, stream>>>(
      ei, E, Et, H1n, ssrc1, sdst1, stage1, m1);
  edge_pass2<<<cdiv((long)Et * H1n, 256), B, 0, stream>>>(
      ei, E, Et, H1n, stage1, m1, z1);
  scatter_msgs<<<cdiv((long)Et * H1n * 32, 256), B, 0, stream>>>(
      ei, E, Et, H1n, D1n, h_t1, stage1, z1, alpha1, acc1);
  // h1 = elu(x@lin1_W + acc1 + b1 + lin1_b)  (overwrites h_t1; old value dead)
  gemm_bf16x3<<<gemmGridW, B, 0, stream>>>(x, l1h, l1l, h_t1,
                                           acc1, b1, lin1_b, N, F, HD1, 1);

  // ================= layer 2 =================
  gemm_bf16x3<<<gemmGridN, B, 0, stream>>>(h_t1, w2h, w2l, h2t,
                                           nullptr, nullptr, nullptr, N, HD1, D2n, 0);
  node_scores<<<cdiv((long)N * 32, 256), B, 0, stream>>>(
      h2t, att_src2, att_dst2, ssrc2, sdst2, N, 1, D2n);
  edge_pass1<<<cdiv((long)Et, 256), B, 0, stream>>>(
      ei, E, Et, 1, ssrc2, sdst2, stage2, m2);
  edge_pass2<<<cdiv((long)Et, 256), B, 0, stream>>>(
      ei, E, Et, 1, stage2, m2, z2);
  scatter_msgs<<<cdiv((long)Et * 32, 256), B, 0, stream>>>(
      ei, E, Et, 1, D2n, h2t, stage2, z2, alpha2, acc2);
  // out = h1@lin2_W + acc2 + b2 + lin2_b  -> d_out
  gemm_bf16x3<<<gemmGridN, B, 0, stream>>>(h_t1, l2h, l2l, out,
                                           acc2, b2, lin2_b, N, HD1, D2n, 2);
}
